// mLSTMLayer_39522289057914
// MI455X (gfx1250) — matrix operand order repaired and checked by reading, not compiled, not run on hardware
//
#include <hip/hip_runtime.h>
#include <hip/hip_bf16.h>

// ---------------------------------------------------------------------------
// mLSTM forward for gfx1250 (MI455X).
//   1) fp32 -> bf16 conversion of x and all large weights (into d_ws)
//   2) WMMA bf16 GEMMs (register-blocked 32x64 per wave): q, k, v, o_preact, y
//      + single-tile WMMA GEMM for the tiny [M,16] gate projection
//   3) soft-cap gate activation
//   4) CHUNKWISE-PARALLEL mLSTM scan (L=64): 4 WMMA GEMMs per chunk per head,
//      sequential depth 16 instead of 1024.  State C^T fp32 in LDS.
//   5) multi-head LayerNorm + sigmoid output gate (emits bf16 for final GEMM)
// ---------------------------------------------------------------------------

typedef __attribute__((ext_vector_type(16))) __bf16 v16bf;
typedef __attribute__((ext_vector_type(8)))  float  v8f;

#define NHEAD 8
#define DQK   64
#define DV    128
#define LCH   64
#define CAPV  15.0f

union BfPack { v16bf v; uint4 u[2]; };

static __device__ __forceinline__ uint4 ldg4(const __bf16* p) {
    return *reinterpret_cast<const uint4*>(p);
}

static __device__ __forceinline__ v8f wmma_bf16(v16bf a, v16bf b, v8f c) {
    return __builtin_amdgcn_wmma_f32_16x16x32_bf16(false, a, false, b, (short)0, c,
                                                   false, false);
}

// A fragment: rows m0..m0+15, contraction cols kc0..kc0+31 (bf16, row-major)
static __device__ __forceinline__ v16bf frag_a(const __bf16* base, int stride,
                                               int m0, int kc0, int half, int l15) {
    const __bf16* p = base + (size_t)(m0 + l15) * stride + kc0 + half * 16;
    BfPack pk;
    pk.u[0] = *reinterpret_cast<const uint4*>(p);
    pk.u[1] = *reinterpret_cast<const uint4*>(p + 8);
    return pk.v;
}
// B fragment: out-cols n0..n0+15 are rows of base; contraction kc0..kc0+31
static __device__ __forceinline__ v16bf frag_b(const __bf16* base, int stride,
                                               int n0, int kc0, int half, int l15) {
    const __bf16* p = base + (size_t)(n0 + l15) * stride + kc0 + half * 16;
    BfPack pk;
    pk.u[0] = *reinterpret_cast<const uint4*>(p);
    pk.u[1] = *reinterpret_cast<const uint4*>(p + 8);
    return pk.v;
}
// B fragment gathered from an fp32 array (converted on the fly)
static __device__ __forceinline__ v16bf frag_b_f32(const float* base, int stride,
                                                   int n0, int kc0, int half, int l15) {
    const float* p = base + (size_t)(n0 + l15) * stride + kc0 + half * 16;
    v16bf r;
#pragma unroll
    for (int e = 0; e < 16; ++e) r[e] = (__bf16)p[e];
    return r;
}

// ---------------------------------------------------------------- conversion
__global__ __launch_bounds__(256)
void cvt_f32_to_bf16(const float* __restrict__ in, __bf16* __restrict__ out, int n) {
    int i = blockIdx.x * 256 + threadIdx.x;
    if (i < n) out[i] = (__bf16)in[i];
}

// ------------------------------------------------- blocked WMMA bf16 GEMM
__global__ __launch_bounds__(256)
void gemm_bf16_wmma_blk(const __bf16* __restrict__ A,
                        const __bf16* __restrict__ W,
                        float* __restrict__ Out,
                        int M, int N, int K)
{
    const int lane = threadIdx.x;
    const int wave = threadIdx.y;
    const int wm = wave >> 1;
    const int wn = wave & 1;
    const int m0 = blockIdx.y * 128 + wm * 32;
    const int n0 = blockIdx.x * 128 + wn * 64;
    const int half = lane >> 4;
    const int l15  = lane & 15;

    const __bf16* arow0 = A + (size_t)(m0 + l15) * K + half * 16;
    const __bf16* arow1 = arow0 + (size_t)16 * K;
    const __bf16* brow = W + (size_t)(n0 + l15) * K + half * 16;
    const size_t bstep = (size_t)16 * K;

    v8f acc[2][4] = {};
    for (int k = 0; k < K; k += 32) {
        BfPack a[2], b[4];
        a[0].u[0] = ldg4(arow0 + k);
        a[0].u[1] = ldg4(arow0 + k + 8);
        a[1].u[0] = ldg4(arow1 + k);
        a[1].u[1] = ldg4(arow1 + k + 8);
#pragma unroll
        for (int j = 0; j < 4; ++j) {
            b[j].u[0] = ldg4(brow + j * bstep + k);
            b[j].u[1] = ldg4(brow + j * bstep + k + 8);
        }
#pragma unroll
        for (int i = 0; i < 2; ++i)
#pragma unroll
            for (int j = 0; j < 4; ++j)
                acc[i][j] = wmma_bf16(a[i].v, b[j].v, acc[i][j]);
    }
#pragma unroll
    for (int i = 0; i < 2; ++i) {
        float* orow = Out + (size_t)(m0 + i * 16 + half * 8) * N + l15;
#pragma unroll
        for (int j = 0; j < 4; ++j) {
            float* op = orow + n0 + j * 16;
#pragma unroll
            for (int r = 0; r < 8; ++r) op[(size_t)r * N] = acc[i][j][r];
        }
    }
}

// --------------------------------------- single-tile WMMA GEMM (small N)
__global__ __launch_bounds__(256)
void gemm_bf16_wmma(const __bf16* __restrict__ A,
                    const __bf16* __restrict__ W,
                    float* __restrict__ Out,
                    int M, int N, int K)
{
    const int lane = threadIdx.x;
    const int wave = threadIdx.y;
    const int m0 = (blockIdx.y * 8 + wave) * 16;
    const int n0 = blockIdx.x * 16;
    if (m0 >= M || n0 >= N) return;
    const int half = lane >> 4;
    const int l15  = lane & 15;
    const __bf16* arow = A + (size_t)(m0 + l15) * K + half * 16;
    const __bf16* brow = W + (size_t)(n0 + l15) * K + half * 16;
    v8f acc = {};
    for (int k = 0; k < K; k += 32) {
        BfPack av, bv;
        av.u[0] = ldg4(arow + k);
        av.u[1] = ldg4(arow + k + 8);
        bv.u[0] = ldg4(brow + k);
        bv.u[1] = ldg4(brow + k + 8);
        acc = wmma_bf16(av.v, bv.v, acc);
    }
    float* orow = Out + (size_t)(m0 + half * 8) * N + n0 + l15;
#pragma unroll
    for (int i = 0; i < 8; ++i) orow[(size_t)i * N] = acc[i];
}

// ----------------------------------------------------- gate preact soft-cap
__global__ __launch_bounds__(256)
void gates_act(const float* __restrict__ gbuf,
               const float* __restrict__ bi, const float* __restrict__ bfg,
               float* __restrict__ ipre, float* __restrict__ fpre, int Mtot)
{
    int idx = blockIdx.x * 256 + threadIdx.x;
    if (idx >= Mtot * NHEAD) return;
    int m = idx >> 3, h = idx & 7;
    float iv = gbuf[m * 16 + h]     + bi[h];
    float fv = gbuf[m * 16 + 8 + h] + bfg[h];
    ipre[idx] = CAPV * tanhf(iv * (1.0f / CAPV));
    fpre[idx] = CAPV * tanhf(fv * (1.0f / CAPV));
}

// --------------------------------------------- chunkwise-parallel mLSTM scan
// One block per (b, head); 256 threads (8 waves).  Dynamic LDS.
__global__ __launch_bounds__(256)
void mlstm_chunk(const float* __restrict__ q, const float* __restrict__ k,
                 const float* __restrict__ v,
                 const float* __restrict__ ipre, const float* __restrict__ fpre,
                 float* __restrict__ hbuf, int S)
{
    extern __shared__ char smem_raw[];
    char* sp = smem_raw;
    float*  Ct    = (float*)sp;  sp += DV * DQK * 4;       // C^T state [v][k] fp32
    __bf16* qsb   = (__bf16*)sp; sp += LCH * DQK * 2;      // q*scale  [tau][k]
    __bf16* kb    = (__bf16*)sp; sp += LCH * DQK * 2;      // k        [sig][k]
    __bf16* vbt   = (__bf16*)sp; sp += DV * LCH * 2;       // v^T      [v][sig]
    __bf16* kpt   = (__bf16*)sp; sp += DQK * LCH * 2;      // (u*k)^T  [k][sig]
    __bf16* Pb    = (__bf16*)sp; sp += LCH * LCH * 2;      // weights  [tau][sig]
    float*  bv    = (float*)sp;  sp += LCH * 4;
    float*  ivv   = (float*)sp;  sp += LCH * 4;
    float*  fvv   = (float*)sp;  sp += LCH * 4;
    float*  mloc  = (float*)sp;  sp += LCH * 4;
    float*  dec0  = (float*)sp;  sp += LCH * 4;
    float*  eneg  = (float*)sp;  sp += LCH * 4;
    float*  uu    = (float*)sp;  sp += LCH * 4;
    float*  denomv= (float*)sp;  sp += LCH * 4;
    float*  nst   = (float*)sp;  sp += DQK * 4;
    float*  mprev = (float*)sp;  sp += 4;

    const int t    = threadIdx.x;
    const int wave = t >> 5;
    const int lane = t & 31;
    const int half = lane >> 4;
    const int l15  = lane & 15;
    const int bh = blockIdx.x, b = bh >> 3, hd = bh & 7;

    for (int i = t; i < DV * DQK; i += 256) Ct[i] = 0.f;
    if (t < DQK) nst[t] = 0.f;
    if (t == 0) *mprev = 0.f;
    __syncthreads();

    const int nchunk = S / LCH;
    for (int c = 0; c < nchunk; ++c) {
        const int s0 = c * LCH;
        // ---- phase 1: stage chunk into LDS (bf16)
        for (int idx = t; idx < LCH * DQK; idx += 256) {
            int sg = idx >> 6, kk = idx & 63;
            size_t off = ((size_t)b * S + s0 + sg) * (NHEAD * DQK) + hd * DQK + kk;
            qsb[idx] = (__bf16)(q[off] * 0.125f);   // 64^-0.5
            kb[idx]  = (__bf16)k[off];
        }
        for (int idx = t; idx < LCH * DV; idx += 256) {
            int sg = idx >> 7, vv = idx & 127;
            size_t off = ((size_t)b * S + s0 + sg) * (NHEAD * DV) + hd * DV + vv;
            vbt[vv * LCH + sg] = (__bf16)v[off];
        }
        if (t < LCH) {
            size_t goff = ((size_t)b * S + s0 + t) * NHEAD + hd;
            ivv[t] = ipre[goff];
            fvv[t] = fpre[goff];
        }
        __syncthreads();
        // ---- phase 2: serial gate prep (exponent args all <= 0: stabilized)
        if (t == 0) {
            float mp = *mprev;
            float accb = 0.f, gmax = -3.0e38f;
            for (int tau = 0; tau < LCH; ++tau) {
                accb += fvv[tau];
                bv[tau] = accb;
                gmax = fmaxf(gmax, ivv[tau] - accb);
                float mm = accb + fmaxf(mp, gmax);
                mloc[tau] = mm;
                dec0[tau] = __expf(mp + accb - mm);
                eneg[tau] = __expf(-mm);
            }
            float blast = bv[LCH - 1], mlast = mloc[LCH - 1];
            for (int sg = 0; sg < LCH; ++sg)
                uu[sg] = __expf(ivv[sg] + blast - bv[sg] - mlast);
            *mprev = mlast;
        }
        __syncthreads();
        // ---- phase 3: inter GEMM (Q_s . C^T) kept in regs; S=Q_s.K^T -> Pb
        v8f accI[4];
#pragma unroll
        for (int j = 0; j < 4; ++j) {               // 32 tiles of [64 x 128]
            int tile = wave * 4 + j;
            int m0 = (tile >> 3) * 16, n0 = (tile & 7) * 16;
            v8f acc = {};
#pragma unroll
            for (int kc = 0; kc < DQK; kc += 32) {
                v16bf a  = frag_a(qsb, DQK, m0, kc, half, l15);
                v16bf bb = frag_b_f32(Ct, DQK, n0, kc, half, l15);
                acc = wmma_bf16(a, bb, acc);
            }
            accI[j] = acc;
        }
#pragma unroll
        for (int j = 0; j < 2; ++j) {               // 16 tiles of [64 x 64]
            int tile = wave * 2 + j;
            int m0 = (tile >> 2) * 16, n0 = (tile & 3) * 16;
            v8f acc = {};
#pragma unroll
            for (int kc = 0; kc < DQK; kc += 32) {
                v16bf a  = frag_a(qsb, DQK, m0, kc, half, l15);
                v16bf bb = frag_b(kb, DQK, n0, kc, half, l15);
                acc = wmma_bf16(a, bb, acc);
            }
            int sg = n0 + l15;
#pragma unroll
            for (int r = 0; r < 8; ++r) {           // mask + decay scale -> P
                int tau = m0 + r + half * 8;
                float wgt = (sg <= tau)
                    ? __expf(ivv[sg] + bv[tau] - bv[sg] - mloc[tau]) : 0.f;
                Pb[tau * LCH + sg] = (__bf16)(acc[r] * wgt);
            }
        }
        __syncthreads();
        // ---- phase 4: qn/denominator (t<64) | (u*k)^T build (t>=64)
        if (t < LCH) {
            float rs = 0.f;
            for (int sg = 0; sg < LCH; ++sg) rs += (float)Pb[t * LCH + sg];
            float qnI = 0.f;
            for (int kk = 0; kk < DQK; ++kk) qnI += (float)qsb[t * DQK + kk] * nst[kk];
            float qn = dec0[t] * qnI + rs;
            denomv[t] = fmaxf(fabsf(qn), eneg[t]) + 1e-6f;
        } else {
            for (int idx = t - 64; idx < LCH * DQK; idx += 192) {
                int sg = idx >> 6, kk = idx & 63;
                kpt[kk * LCH + sg] = (__bf16)(uu[sg] * (float)kb[sg * DQK + kk]);
            }
        }
        __syncthreads();
        // ---- phase 5: H = diag(dec0).H_inter + P.V ; write h to global
#pragma unroll
        for (int j = 0; j < 4; ++j) {
            int tile = wave * 4 + j;
            int m0 = (tile >> 3) * 16, n0 = (tile & 7) * 16;
            v8f acc = accI[j];
#pragma unroll
            for (int r = 0; r < 8; ++r) acc[r] *= dec0[m0 + r + half * 8];
#pragma unroll
            for (int kc = 0; kc < LCH; kc += 32) {
                v16bf a  = frag_a(Pb, LCH, m0, kc, half, l15);
                v16bf bb = frag_b(vbt, LCH, n0, kc, half, l15);
                acc = wmma_bf16(a, bb, acc);
            }
#pragma unroll
            for (int r = 0; r < 8; ++r) {
                int tau = m0 + r + half * 8;
                size_t off = ((size_t)b * S + s0 + tau) * (NHEAD * DV) + hd * DV + n0 + l15;
                hbuf[off] = acc[r] / denomv[tau];
            }
        }
        // ---- phase 6: state update  C^T = dec_last*C^T + V^T.(u*k) ; n update
        {
            const float d0l = dec0[LCH - 1];
#pragma unroll
            for (int j = 0; j < 4; ++j) {           // 32 tiles of [128 x 64]
                int tile = wave * 4 + j;
                int m0 = (tile >> 2) * 16, n0 = (tile & 3) * 16;
                v8f acc;
#pragma unroll
                for (int r = 0; r < 8; ++r)
                    acc[r] = d0l * Ct[(m0 + r + half * 8) * DQK + n0 + l15];
#pragma unroll
                for (int kc = 0; kc < LCH; kc += 32) {
                    v16bf a  = frag_a(vbt, LCH, m0, kc, half, l15);
                    v16bf bb = frag_b(kpt, LCH, n0, kc, half, l15);
                    acc = wmma_bf16(a, bb, acc);
                }
#pragma unroll
                for (int r = 0; r < 8; ++r)
                    Ct[(m0 + r + half * 8) * DQK + n0 + l15] = acc[r];
            }
            if (t < DQK) {
                float nn = d0l * nst[t];
                for (int sg = 0; sg < LCH; ++sg)
                    nn += uu[sg] * (float)kb[sg * DQK + t];
                nst[t] = nn;
            }
        }
        __syncthreads();
    }
}

// -------------------------------------- LayerNorm + sigmoid gate -> bf16
__global__ __launch_bounds__(256)
void ln_gate(const float* __restrict__ hbuf, const float* __restrict__ opre,
             const float* __restrict__ ln_w, const float* __restrict__ ln_b,
             __bf16* __restrict__ hout, int Dv)
{
    const int m = blockIdx.x;
    const int t = threadIdx.x;
    __shared__ float red[256];

    const float* row = hbuf + (size_t)m * Dv;
    float sum = 0.f, sumsq = 0.f;
    for (int i = t; i < Dv; i += 256) {
        float x = row[i];
        sum += x; sumsq += x * x;
    }
    red[t] = sum; __syncthreads();
    for (int off = 128; off > 0; off >>= 1) {
        if (t < off) red[t] += red[t + off];
        __syncthreads();
    }
    const float mu = red[0] / Dv;
    __syncthreads();
    red[t] = sumsq; __syncthreads();
    for (int off = 128; off > 0; off >>= 1) {
        if (t < off) red[t] += red[t + off];
        __syncthreads();
    }
    const float var  = red[0] / Dv - mu * mu;
    const float rstd = rsqrtf(var + 1e-6f);

    const float* og = opre + (size_t)m * Dv;
    __bf16* orow = hout + (size_t)m * Dv;
    for (int i = t; i < Dv; i += 256) {
        float hn = (row[i] - mu) * rstd * ln_w[i] + ln_b[i];
        float sg = 1.f / (1.f + __expf(-og[i]));
        orow[i] = (__bf16)(sg * hn);
    }
}

// ---------------------------------------------------------------------------
extern "C" void kernel_launch(void* const* d_in, const int* in_sizes, int n_in,
                              void* d_out, int out_size, void* d_ws, size_t ws_size,
                              hipStream_t stream) {
    const int B = 2, S = 1024, D = 1024, QK = 512, Vd = 1024;
    const int M = B * S;

    const float* x    = (const float*)d_in[0];
    const float* Wq   = (const float*)d_in[1];
    const float* Wk   = (const float*)d_in[2];
    const float* Wv   = (const float*)d_in[3];
    const float* Wog  = (const float*)d_in[4];
    const float* Wi   = (const float*)d_in[5];
    const float* bi   = (const float*)d_in[6];
    const float* Wf   = (const float*)d_in[7];
    const float* bfg  = (const float*)d_in[8];
    const float* ln_w = (const float*)d_in[9];
    const float* ln_b = (const float*)d_in[10];
    const float* Wout = (const float*)d_in[11];
    float* y = (float*)d_out;

    char* ws = (char*)d_ws;
    auto carve = [&](size_t bytes) {
        char* p = ws;
        ws += (bytes + 255) & ~(size_t)255;
        return p;
    };
    __bf16* xb    = (__bf16*)carve((size_t)M * D * 2);
    __bf16* wq_b  = (__bf16*)carve((size_t)QK * D * 2);
    __bf16* wk_b  = (__bf16*)carve((size_t)QK * D * 2);
    __bf16* wv_b  = (__bf16*)carve((size_t)Vd * D * 2);
    __bf16* wog_b = (__bf16*)carve((size_t)Vd * D * 2);
    __bf16* wout_b= (__bf16*)carve((size_t)D * Vd * 2);
    __bf16* wif_b = (__bf16*)carve((size_t)16 * D * 2);
    float*  qf    = (float*)carve((size_t)M * QK * 4);
    float*  kf    = (float*)carve((size_t)M * QK * 4);
    float*  vf    = (float*)carve((size_t)M * Vd * 4);
    float*  ogf   = (float*)carve((size_t)M * Vd * 4);
    float*  gbuf  = (float*)carve((size_t)M * 16 * 4);
    float*  ipre  = (float*)carve((size_t)M * NHEAD * 4);
    float*  fpre  = (float*)carve((size_t)M * NHEAD * 4);
    float*  hbuf  = (float*)carve((size_t)M * Vd * 4);
    __bf16* houtb = (__bf16*)carve((size_t)M * Vd * 2);

    auto cvt = [&](const float* src, __bf16* dst, int n) {
        cvt_f32_to_bf16<<<dim3((n + 255) / 256), dim3(256), 0, stream>>>(src, dst, n);
    };
    cvt(x,    xb,     M * D);
    cvt(Wq,   wq_b,   QK * D);
    cvt(Wk,   wk_b,   QK * D);
    cvt(Wv,   wv_b,   Vd * D);
    cvt(Wog,  wog_b,  Vd * D);
    cvt(Wout, wout_b, D * Vd);
    cvt(Wi,   wif_b,            NHEAD * D);
    cvt(Wf,   wif_b + NHEAD*D,  NHEAD * D);

    auto gemm = [&](const __bf16* A, const __bf16* W, float* Out, int Mm, int N, int K) {
        if ((N % 128) == 0 && (Mm % 128) == 0) {
            gemm_bf16_wmma_blk<<<dim3(N / 128, Mm / 128), dim3(32, 8), 0, stream>>>(
                A, W, Out, Mm, N, K);
        } else {
            gemm_bf16_wmma<<<dim3((N + 15) / 16, (Mm + 127) / 128), dim3(32, 8), 0, stream>>>(
                A, W, Out, Mm, N, K);
        }
    };
    gemm(xb, wq_b,  qf,   M, QK, D);
    gemm(xb, wk_b,  kf,   M, QK, D);
    gemm(xb, wv_b,  vf,   M, Vd, D);
    gemm(xb, wog_b, ogf,  M, Vd, D);
    gemm(xb, wif_b, gbuf, M, 16, D);

    gates_act<<<dim3((M * NHEAD + 255) / 256), dim3(256), 0, stream>>>(
        gbuf, bi, bfg, ipre, fpre, M);

    // dynamic LDS: Ct(32K) + qsb/kb(8K+8K) + vbt(16K) + kpt(8K) + Pb(8K) + vecs
    const size_t smem = (size_t)(DV*DQK*4 + LCH*DQK*2*2 + DV*LCH*2 + DQK*LCH*2 +
                                 LCH*LCH*2 + 8*LCH*4 + DQK*4 + 256);
    mlstm_chunk<<<dim3(B * NHEAD), dim3(256), smem, stream>>>(
        qf, kf, vf, ipre, fpre, hbuf, S);

    ln_gate<<<dim3(M), dim3(256), 0, stream>>>(
        hbuf, ogf, ln_w, ln_b, houtb, Vd);

    gemm(houtb, wout_b, y, M, D, Vd);

    (void)in_sizes; (void)n_in; (void)out_size; (void)ws_size;
}
